// MultiHeadAttention_22239340658988
// MI455X (gfx1250) — matrix operand order repaired and checked by reading, not compiled, not run on hardware
//
#include <hip/hip_runtime.h>

typedef _Float16 half_t;
typedef __attribute__((ext_vector_type(4)))  _Float16 v4h;
typedef __attribute__((ext_vector_type(8)))  _Float16 v8h;
typedef __attribute__((ext_vector_type(16))) _Float16 v16h;
typedef __attribute__((ext_vector_type(4)))  float    v4f;
typedef __attribute__((ext_vector_type(8)))  float    v8f;
typedef __attribute__((ext_vector_type(4)))  unsigned int v4u;
typedef __attribute__((ext_vector_type(4)))  int          v4i;
typedef __attribute__((ext_vector_type(8)))  int          v8i;

#define BATCH 4
#define SEQ   2048
#define EMB   1024
#define HEADS 16
#define HDIM  64

// ---- Tensor Data Mover availability (compile-safe gating) -----------------
#if defined(__has_builtin)
# if __has_builtin(__builtin_amdgcn_tensor_load_to_lds) && \
     __has_builtin(__builtin_amdgcn_s_wait_tensorcnt)
#  define USE_TDM 1
# endif
#endif
#ifndef USE_TDM
# define USE_TDM 0
#endif

__device__ __forceinline__ v16h cat16(v8h lo, v8h hi) {
  return __builtin_shufflevector(lo, hi, 0,1,2,3,4,5,6,7,8,9,10,11,12,13,14,15);
}

// 16-lane butterfly reductions via ds_swizzle (group-of-32 mode: and=0x1F, xor in [14:10])
__device__ __forceinline__ float redmax16(float x) {
  x = fmaxf(x, __int_as_float(__builtin_amdgcn_ds_swizzle(__float_as_int(x), 0x041F)));
  x = fmaxf(x, __int_as_float(__builtin_amdgcn_ds_swizzle(__float_as_int(x), 0x081F)));
  x = fmaxf(x, __int_as_float(__builtin_amdgcn_ds_swizzle(__float_as_int(x), 0x101F)));
  x = fmaxf(x, __int_as_float(__builtin_amdgcn_ds_swizzle(__float_as_int(x), 0x201F)));
  return x;
}
__device__ __forceinline__ float redsum16(float x) {
  x += __int_as_float(__builtin_amdgcn_ds_swizzle(__float_as_int(x), 0x041F));
  x += __int_as_float(__builtin_amdgcn_ds_swizzle(__float_as_int(x), 0x081F));
  x += __int_as_float(__builtin_amdgcn_ds_swizzle(__float_as_int(x), 0x101F));
  x += __int_as_float(__builtin_amdgcn_ds_swizzle(__float_as_int(x), 0x201F));
  return x;
}

#if USE_TDM
// DMA one 2D f16 tile (64 rows x 64 elements) from global into LDS.
// Hardware pad: 4 dwords after every 32 dwords -> LDS row stride 72 halves.
// D# layout per CDNA5 ISA ch.8 (group0: count/lds/global/type; group1: dims).
__device__ __forceinline__ void tdm_load_tile64x64(unsigned int lds_off,
                                                   const half_t* g,
                                                   unsigned int dim0,
                                                   unsigned long long stride0) {
  unsigned long long ga = (unsigned long long)(uintptr_t)g;
  v4u g0;
  g0[0] = 1u;                                    // count=1 (valid descriptor)
  g0[1] = lds_off;                               // lds_addr (bytes)
  g0[2] = (unsigned int)ga;                      // global_addr[31:0]
  g0[3] = (unsigned int)(ga >> 32) | (2u << 30); // global_addr[56:32] | type=2
  v8i g1;
  // data_size=2B (1<<16) | pad_enable (1<<20) | pad_interval=32dw (4<<22)
  // | pad_amount=4dw (3<<25)
  g1[0] = (int)((1u << 16) | (1u << 20) | (4u << 22) | (3u << 25));
  g1[1] = (int)((dim0 & 0xFFFFu) << 16);               // tensor_dim0[15:0]
  g1[2] = (int)((dim0 >> 16) | (64u << 16));           // tensor_dim0[31:16] | tensor_dim1[15:0]=64
  g1[3] = (int)(64u << 16);                            // tensor_dim1[31:16]=0 | tile_dim0=64
  g1[4] = (int)64u;                                    // tile_dim1=64 | tile_dim2=0
  g1[5] = (int)(unsigned int)stride0;                  // tensor_dim0_stride[31:0]
  g1[6] = (int)(unsigned int)(stride0 >> 32);          // stride[47:32] | dim1_stride lo = 0
  g1[7] = 0;
  v4i z4 = (v4i){};
#if __clang_major__ >= 23
  v8i z8 = (v8i){};
  __builtin_amdgcn_tensor_load_to_lds(g0, g1, z4, z4, z8, 0);
#else
  __builtin_amdgcn_tensor_load_to_lds(g0, g1, z4, z4, 0);
#endif
}
#endif  // USE_TDM

// ---------------------------------------------------------------------------
// Kernel 1: out = X @ W^T + b, written as f16.
//   vtrans == 0 : [B, H, S, D]   (Q, K)
//   vtrans == 1 : [B, H, D, S]   (V, pre-transposed for the PV B-fragments)
// Block = 256 thr (8 waves), tile 128(M) x 128(N), K-step 32 (one WMMA).
// Two-stage software pipeline: global loads for slice i+1 are issued before
// the WMMA work of slice i; cvt+LDS-store lands behind the compute.
// ---------------------------------------------------------------------------
#define LDA 40  // 32 + 8 halves padding -> conflict-free 16B fragment reads

__global__ __launch_bounds__(256)
void proj_kernel(const float* __restrict__ X, const float* __restrict__ W,
                 const float* __restrict__ bias, half_t* __restrict__ dst,
                 int vtrans) {
  __shared__ half_t As[2][128 * LDA];
  __shared__ half_t Bs[2][128 * LDA];

  const int tid  = threadIdx.x;
  const int lane = tid & 31;
  const int w    = tid >> 5;
  const int hi   = (lane >> 4) & 1;
  const int l15  = lane & 15;
  const int wm   = w >> 1;          // 0..3  (M position, 32 rows each)
  const int wn   = w & 1;           // 0..1  (N position, 64 cols each)
  const int Mbase = blockIdx.y * 128;
  const int Nbase = blockIdx.x * 128;

  // staging coordinates for this thread (4 chunks of 4 floats)
  int srow[4], scol[4];
#pragma unroll
  for (int i = 0; i < 4; ++i) {
    int c = tid + i * 256;          // 0..1023
    srow[i] = c >> 3;               // 0..127
    scol[i] = (c & 7) * 4;          // 0..28
  }

  v8f acc[2][4];
#pragma unroll
  for (int i = 0; i < 2; ++i)
#pragma unroll
    for (int j = 0; j < 4; ++j) acc[i][j] = (v8f){};

  v4f xa[4], wbv[4];
  // prologue: fetch slice 0 and stage it into buffer 0
#pragma unroll
  for (int i = 0; i < 4; ++i) {
    xa[i]  = *(const v4f*)(X + (size_t)(Mbase + srow[i]) * EMB + scol[i]);
    wbv[i] = *(const v4f*)(W + (size_t)(Nbase + srow[i]) * EMB + scol[i]);
  }
#pragma unroll
  for (int i = 0; i < 4; ++i) {
    v4h xh, wh;
#pragma unroll
    for (int e = 0; e < 4; ++e) { xh[e] = (half_t)xa[i][e]; wh[e] = (half_t)wbv[i][e]; }
    *(v4h*)&As[0][srow[i] * LDA + scol[i]] = xh;
    *(v4h*)&Bs[0][srow[i] * LDA + scol[i]] = wh;
  }

  for (int kk = 0; kk < EMB; kk += 32) {
    const int buf = (kk >> 5) & 1;
    const int more = (kk + 32 < EMB);
    __syncthreads();   // slice kk staged in As/Bs[buf]; reads of [buf^1] done

    // issue global loads for slice kk+32 (latency hidden behind WMMAs below)
    if (more) {
#pragma unroll
      for (int i = 0; i < 4; ++i) {
        xa[i]  = *(const v4f*)(X + (size_t)(Mbase + srow[i]) * EMB + kk + 32 + scol[i]);
        wbv[i] = *(const v4f*)(W + (size_t)(Nbase + srow[i]) * EMB + kk + 32 + scol[i]);
      }
    }

    // fragments + 8 WMMAs from buffer `buf`
    v16h a[2], b[4];
#pragma unroll
    for (int i = 0; i < 2; ++i) {
      const half_t* p = &As[buf][(wm * 32 + i * 16 + l15) * LDA + hi * 8];
      a[i] = cat16(*(const v8h*)p, *(const v8h*)(p + 16));
    }
#pragma unroll
    for (int j = 0; j < 4; ++j) {
      const half_t* p = &Bs[buf][(wn * 64 + j * 16 + l15) * LDA + hi * 8];
      b[j] = cat16(*(const v8h*)p, *(const v8h*)(p + 16));
    }
#pragma unroll
    for (int i = 0; i < 2; ++i)
#pragma unroll
      for (int j = 0; j < 4; ++j)
        acc[i][j] = __builtin_amdgcn_wmma_f32_16x16x32_f16(
            false, a[i], false, b[j], (short)0, acc[i][j], false, false);

    // stage slice kk+32 into the alternate buffer
    if (more) {
#pragma unroll
      for (int i = 0; i < 4; ++i) {
        v4h xh, wh;
#pragma unroll
        for (int e = 0; e < 4; ++e) { xh[e] = (half_t)xa[i][e]; wh[e] = (half_t)wbv[i][e]; }
        *(v4h*)&As[buf ^ 1][srow[i] * LDA + scol[i]] = xh;
        *(v4h*)&Bs[buf ^ 1][srow[i] * LDA + scol[i]] = wh;
      }
    }
  }

  // epilogue: bias add, f16 store
#pragma unroll
  for (int i = 0; i < 2; ++i) {
#pragma unroll
    for (int j = 0; j < 4; ++j) {
      int n = Nbase + wn * 64 + j * 16 + l15;
      float bv = bias[n];
      int h = n >> 6, d = n & 63;
#pragma unroll
      for (int r = 0; r < 8; ++r) {
        int m = Mbase + wm * 32 + i * 16 + r + hi * 8;
        int bidx = m >> 11;             // m / SEQ
        int s    = m & (SEQ - 1);
        half_t val = (half_t)(acc[i][j][r] + bv);
        if (vtrans)
          dst[(((size_t)(bidx * HEADS + h)) * HDIM + d) * SEQ + s] = val;
        else
          dst[(((size_t)(bidx * HEADS + h)) * SEQ + s) * HDIM + d] = val;
      }
    }
  }
}

// ---------------------------------------------------------------------------
// Kernel 2: flash attention.  Block = 128 thr (4 waves); each wave owns 16
// queries; block covers 64 queries of one (b,h).  K tiles [key][d] and
// pre-transposed V tiles [d][key] are staged into double-buffered LDS by the
// Tensor Data Mover: wave 0 waits (s_wait_tensorcnt) for tile i, the block
// barriers, then wave 0 issues the DMA for tile i+1 into the alternate
// buffer while all waves compute tile i -- the DMA is fully hidden behind
// 16 WMMAs + softmax.  Fallback: vectorized LDS copies.
// ---------------------------------------------------------------------------
#define LDK 72  // 64 + 8 halves padding (matches TDM pad: 4dw per 32dw)

__global__ __launch_bounds__(128)
void attn_kernel(const half_t* __restrict__ q16, const half_t* __restrict__ k16,
                 const half_t* __restrict__ vT, float* __restrict__ out) {
  __shared__ half_t Ks[2][64 * LDK];
  __shared__ half_t Vs[2][64 * LDK];
  __shared__ half_t Ps[4 * 16 * LDK];

  const int tid  = threadIdx.x;
  const int lane = tid & 31;
  const int w    = tid >> 5;       // 0..3
  const int hi   = (lane >> 4) & 1;
  const int l15  = lane & 15;
  const int bh   = blockIdx.y;     // b*HEADS + h
  const int qblk = blockIdx.x;     // 0..31
  const size_t headoff = (size_t)bh * SEQ * HDIM;   // same for [S,D] and [D,S]
  const int qrow = qblk * 64 + w * 16;
  half_t* myP = &Ps[w * 16 * LDK];

  // Q A-fragments, resident across the whole key loop (K-tiles d=0..31, 32..63)
  v16h aq[2];
#pragma unroll
  for (int t = 0; t < 2; ++t) {
    const half_t* p = q16 + headoff + (size_t)(qrow + l15) * HDIM + t * 32 + hi * 8;
    aq[t] = cat16(*(const v8h*)p, *(const v8h*)(p + 16));
  }

  v8f o[4];
#pragma unroll
  for (int jd = 0; jd < 4; ++jd) o[jd] = (v8f){};
  v8f mrun, lrun;
#pragma unroll
  for (int r = 0; r < 8; ++r) { mrun[r] = -1.0e30f; lrun[r] = 0.0f; }

#if USE_TDM
  // prologue: DMA tile 0 into buffer 0
  if (w == 0) {
    tdm_load_tile64x64((unsigned int)(uintptr_t)&Ks[0][0],
                       k16 + headoff, /*dim0*/HDIM, /*stride0*/HDIM);
    tdm_load_tile64x64((unsigned int)(uintptr_t)&Vs[0][0],
                       vT + headoff, /*dim0*/SEQ, /*stride0*/SEQ);
  }
#endif

  for (int kt = 0; kt < SEQ; kt += 64) {
    const int buf = (kt >> 6) & 1;
#if USE_TDM
    if (w == 0) __builtin_amdgcn_s_wait_tensorcnt(0);  // tile kt arrived
    __syncthreads();                                   // publish; old reads done
    if (w == 0 && kt + 64 < SEQ) {
      // prefetch tile kt+64 into the alternate buffer (hidden behind compute)
      tdm_load_tile64x64((unsigned int)(uintptr_t)&Ks[buf ^ 1][0],
                         k16 + headoff + (size_t)(kt + 64) * HDIM, HDIM, HDIM);
      tdm_load_tile64x64((unsigned int)(uintptr_t)&Vs[buf ^ 1][0],
                         vT + headoff + kt + 64, SEQ, SEQ);
    }
#else
    __syncthreads();
#pragma unroll
    for (int i = 0; i < 4; ++i) {
      int c   = tid + i * 128;       // 0..511
      int row = c >> 3;              // 0..63
      int col = (c & 7) * 8;         // 0..56
      *(v8h*)&Ks[buf][row * LDK + col] =
          *(const v8h*)(k16 + headoff + (size_t)(kt + row) * HDIM + col);
      *(v8h*)&Vs[buf][row * LDK + col] =
          *(const v8h*)(vT + headoff + (size_t)row * SEQ + kt + col);
    }
    __syncthreads();
#endif

    // scores for 4 key sub-tiles of 16
    v8f sc[4];
#pragma unroll
    for (int j = 0; j < 4; ++j) {
      v8f c = (v8f){};
#pragma unroll
      for (int t = 0; t < 2; ++t) {
        const half_t* p = &Ks[buf][(j * 16 + l15) * LDK + t * 32 + hi * 8];
        v16h bk = cat16(*(const v8h*)p, *(const v8h*)(p + 16));
        c = __builtin_amdgcn_wmma_f32_16x16x32_f16(
            false, aq[t], false, bk, (short)0, c, false, false);
      }
#pragma unroll
      for (int r = 0; r < 8; ++r) sc[j][r] = floorf(c[r] * 0.125f);
    }

    // online softmax (rows live at VGPR slot r, lane-half hi; cols across 16 lanes)
    v8f mnew, alpha;
#pragma unroll
    for (int r = 0; r < 8; ++r) {
      float mx = fmaxf(fmaxf(sc[0][r], sc[1][r]), fmaxf(sc[2][r], sc[3][r]));
      mx = redmax16(mx);
      mnew[r]  = fmaxf(mrun[r], mx);
      alpha[r] = __expf(mrun[r] - mnew[r]);
    }
#pragma unroll
    for (int r = 0; r < 8; ++r) {
      float s4 = 0.0f;
#pragma unroll
      for (int j = 0; j < 4; ++j) {
        float p = __expf(sc[j][r] - mnew[r]);
        sc[j][r] = p;
        s4 += p;
      }
      lrun[r] = lrun[r] * alpha[r] + redsum16(s4);
      mrun[r] = mnew[r];
    }
#pragma unroll
    for (int jd = 0; jd < 4; ++jd)
#pragma unroll
      for (int r = 0; r < 8; ++r) o[jd][r] *= alpha[r];

    // P: D-layout -> LDS -> A-layout (per-wave region; same-wave LDS is in-order)
#pragma unroll
    for (int j = 0; j < 4; ++j)
#pragma unroll
      for (int r = 0; r < 8; ++r)
        myP[(r + hi * 8) * LDK + j * 16 + l15] = (half_t)sc[j][r];
    __asm__ volatile("" ::: "memory");
    __builtin_amdgcn_wave_barrier();

    v16h ap[2];
#pragma unroll
    for (int t = 0; t < 2; ++t) {
      const half_t* p = &myP[l15 * LDK + t * 32 + hi * 8];
      ap[t] = cat16(*(const v8h*)p, *(const v8h*)(p + 16));
    }
    // O += P @ V   (B-frags straight from the pre-transposed V tile)
#pragma unroll
    for (int jd = 0; jd < 4; ++jd)
#pragma unroll
      for (int t = 0; t < 2; ++t) {
        const half_t* p = &Vs[buf][(jd * 16 + l15) * LDK + t * 32 + hi * 8];
        v16h bv = cat16(*(const v8h*)p, *(const v8h*)(p + 16));
        o[jd] = __builtin_amdgcn_wmma_f32_16x16x32_f16(
            false, ap[t], false, bv, (short)0, o[jd], false, false);
      }
  }

  // normalize and store f32 output [B, S, H*D]
  const int bidx = bh >> 4;
  const int h    = bh & 15;
#pragma unroll
  for (int jd = 0; jd < 4; ++jd) {
    int d = jd * 16 + l15;
#pragma unroll
    for (int r = 0; r < 8; ++r) {
      int s = qrow + r + hi * 8;
      float inv = 1.0f / lrun[r];
      out[((size_t)(bidx * SEQ + s)) * EMB + h * HDIM + d] = o[jd][r] * inv;
    }
  }
}

// ---------------------------------------------------------------------------
extern "C" void kernel_launch(void* const* d_in, const int* in_sizes, int n_in,
                              void* d_out, int out_size, void* d_ws, size_t ws_size,
                              hipStream_t stream) {
  (void)in_sizes; (void)n_in; (void)out_size; (void)ws_size;
  const float* key   = (const float*)d_in[0];
  const float* query = (const float*)d_in[1];
  const float* value = (const float*)d_in[2];
  const float* Wq    = (const float*)d_in[3];
  const float* bq    = (const float*)d_in[4];
  const float* Wk    = (const float*)d_in[5];
  const float* bk    = (const float*)d_in[6];
  const float* Wv    = (const float*)d_in[7];
  const float* bv    = (const float*)d_in[8];
  float* out = (float*)d_out;

  const size_t BHSD = (size_t)BATCH * HEADS * SEQ * HDIM;  // 8,388,608 elems
  half_t* q16 = (half_t*)d_ws;
  half_t* k16 = q16 + BHSD;
  half_t* vT  = k16 + BHSD;   // [B, H, D, S]

  dim3 gproj(EMB / 128, (BATCH * SEQ) / 128, 1);  // (8, 64)
  proj_kernel<<<gproj, 256, 0, stream>>>(query, Wq, bq, q16, 0);
  proj_kernel<<<gproj, 256, 0, stream>>>(key,   Wk, bk, k16, 0);
  proj_kernel<<<gproj, 256, 0, stream>>>(value, Wv, bv, vT,  1);

  dim3 gattn(SEQ / 64, BATCH * HEADS, 1);         // (32, 64)
  attn_kernel<<<gattn, 128, 0, stream>>>(q16, k16, vT, out);
}
